// Attention_56006373540507
// MI455X (gfx1250) — matrix operand order repaired and checked by reading, not compiled, not run on hardware
//
#include <hip/hip_runtime.h>
#include <hip/hip_bf16.h>

// Problem constants (fixed by the reference)
#define BB 4
#define TT 2048
#define CC 1024
#define HH 16
#define DD 64
#define SCALE 0.125f   // D^-0.5 = 64^-0.5

typedef __attribute__((ext_vector_type(16))) _Float16 v16h;
typedef __attribute__((ext_vector_type(8)))  _Float16 v8h;
typedef __attribute__((ext_vector_type(8)))  float    v8f;

#define WMMA_F16(a, b, c) \
  __builtin_amdgcn_wmma_f32_16x16x32_f16(false, (a), false, (b), (short)0, (c), false, false)

__device__ __forceinline__ v16h cat8(v8h lo, v8h hi) {
  return __builtin_shufflevector(lo, hi, 0, 1, 2, 3, 4, 5, 6, 7, 8, 9, 10, 11, 12, 13, 14, 15);
}

// A-operand fragment (16x32 f16, row-major source, ld in halfs).
// Lane L: M = L&15, h = L>>4; halfs 0..7 -> K = k0 + h*8 + i ; halfs 8..15 -> K = k0 + 16 + h*8 + i.
__device__ __forceinline__ v16h load_a_frag(const _Float16* base, int ld, int row, int k0, int hsel) {
  const _Float16* p = base + (size_t)row * ld + k0 + hsel * 8;
  v8h lo = *(const v8h*)(p);
  v8h hi = *(const v8h*)(p + 16);
  return cat8(lo, hi);
}

// B-operand fragment (32x16 f16). Source is B^T stored N x K row-major (ld in halfs).
// Lane L: N = L&15, h = L>>4; halfs 0..7 -> K = k0 + h*8 + i ; halfs 8..15 -> K = k0 + 16 + h*8 + i.
__device__ __forceinline__ v16h load_b_frag(const _Float16* base, int ld, int nrow, int k0, int hsel) {
  const _Float16* p = base + (size_t)nrow * ld + k0 + hsel * 8;
  v8h lo = *(const v8h*)(p);
  v8h hi = *(const v8h*)(p + 16);
  return cat8(lo, hi);
}

// ---------------------------------------------------------------------------
// Stage 0a: f32 -> f16 elementwise convert
__global__ __launch_bounds__(256) void cvt_f32_to_f16_kernel(
    const float* __restrict__ in, _Float16* __restrict__ out, int n) {
  int i = blockIdx.x * blockDim.x + threadIdx.x;
  if (i < n) out[i] = (_Float16)in[i];
}

// Stage 0b: f32 [rows x cols] -> f16 transposed [cols x rows] (gives B^T layout for WMMA)
__global__ __launch_bounds__(256) void transpose_f32_to_f16_kernel(
    const float* __restrict__ in, _Float16* __restrict__ out, int rows, int cols) {
  int i = blockIdx.x * blockDim.x + threadIdx.x;
  if (i < rows * cols) {
    int r = i / cols, c = i % cols;
    out[(size_t)c * rows + r] = (_Float16)in[i];
  }
}

// ---------------------------------------------------------------------------
// C/D-tile store helper. C layout: row = r + 8*hsel, col = lane&15.
// mode 0/1: dstH in [B,H,T,D] ; mode 2: dstH in [B,H,D,T] ; mode 3: f32 [M,N]+bias
__device__ __forceinline__ void store_ctile(
    v8f c, int mBase, int nBase, int l15, int hsel,
    _Float16* __restrict__ dstH, float* __restrict__ dstF,
    const float* __restrict__ bias, int N, int mode) {
#pragma unroll
  for (int r = 0; r < 8; ++r) {
    int m = mBase + r + 8 * hsel;
    int n = nBase + l15;
    float val = c[r];
    if (mode == 3) {
      dstF[(size_t)m * N + n] = val + bias[n];
    } else {
      int bb = m >> 11, t = m & (TT - 1);       // TT = 2048
      int h = n >> 6, d = n & 63;
      size_t idx = (mode == 2)
                 ? ((((size_t)(bb * HH + h)) * DD + d) * TT + t)    // [B,H,D,T]
                 : ((((size_t)(bb * HH + h)) * TT + t) * DD + d);   // [B,H,T,D]
      dstH[idx] = (_Float16)val;
    }
  }
}

// ---------------------------------------------------------------------------
// Register-blocked WMMA GEMM: each wave computes a 32x32 output tile
// (2x2 WMMA tiles -> 4 wmma per k-step for only 4 fragment loads).
__global__ __launch_bounds__(256) void gemm_wmma_kernel(
    const _Float16* __restrict__ A, const _Float16* __restrict__ Bt,
    _Float16* __restrict__ dstH, float* __restrict__ dstF,
    const float* __restrict__ bias, int M, int N, int K, int mode) {
  int lane = threadIdx.x & 31;
  int l15  = lane & 15;
  int hsel = lane >> 4;
  int gw   = (blockIdx.x * blockDim.x + threadIdx.x) >> 5;
  int nTiles = N >> 5;                    // 32-wide N tiles
  int mT = gw / nTiles;
  int nT = gw - mT * nTiles;
  if (mT >= (M >> 5)) return;

  int aRow0 = mT * 32 + l15, aRow1 = aRow0 + 16;
  int bRow0 = nT * 32 + l15, bRow1 = bRow0 + 16;

  v8f c00 = {}, c01 = {}, c10 = {}, c11 = {};
  for (int kk = 0; kk < K; kk += 32) {
    v16h a0 = load_a_frag(A,  K, aRow0, kk, hsel);
    v16h a1 = load_a_frag(A,  K, aRow1, kk, hsel);
    v16h b0 = load_b_frag(Bt, K, bRow0, kk, hsel);
    v16h b1 = load_b_frag(Bt, K, bRow1, kk, hsel);
    c00 = WMMA_F16(a0, b0, c00);
    c01 = WMMA_F16(a0, b1, c01);
    c10 = WMMA_F16(a1, b0, c10);
    c11 = WMMA_F16(a1, b1, c11);
    if (kk + 32 < K) {                    // hide L2 latency for the next k-tile
      __builtin_prefetch(A  + (size_t)aRow0 * K + kk + 32, 0, 1);
      __builtin_prefetch(A  + (size_t)aRow1 * K + kk + 32, 0, 1);
      __builtin_prefetch(Bt + (size_t)bRow0 * K + kk + 32, 0, 1);
      __builtin_prefetch(Bt + (size_t)bRow1 * K + kk + 32, 0, 1);
    }
  }

  store_ctile(c00, mT * 32 +  0, nT * 32 +  0, l15, hsel, dstH, dstF, bias, N, mode);
  store_ctile(c01, mT * 32 +  0, nT * 32 + 16, l15, hsel, dstH, dstF, bias, N, mode);
  store_ctile(c10, mT * 32 + 16, nT * 32 +  0, l15, hsel, dstH, dstF, bias, N, mode);
  store_ctile(c11, mT * 32 + 16, nT * 32 + 16, l15, hsel, dstH, dstF, bias, N, mode);
}

// ---------------------------------------------------------------------------
// Flash attention: one wave handles 16 queries of one (b,h); streams keys in
// blocks of 32; online softmax in f32; P goes through LDS to convert the
// C-layout scores into an A-layout fragment for the P*V WMMA.
__global__ __launch_bounds__(256) void attn_flash_kernel(
    const _Float16* __restrict__ Qf, const _Float16* __restrict__ Kf,
    const _Float16* __restrict__ Vt, const int* __restrict__ amask,
    _Float16* __restrict__ AO) {
  __shared__ __align__(16) _Float16 smem[8][16 * 32];

  int lane = threadIdx.x & 31;
  int l15  = lane & 15;
  int hsel = lane >> 4;
  int wv   = threadIdx.x >> 5;
  int gw   = blockIdx.x * 8 + wv;

  const int qTiles = TT / 16;                 // 128
  int qT = gw % qTiles;
  int hh = (gw / qTiles) % HH;
  int b  = gw / (qTiles * HH);

  const _Float16* Qbh = Qf + ((size_t)(b * HH + hh)) * TT * DD;
  const _Float16* Kbh = Kf + ((size_t)(b * HH + hh)) * TT * DD;
  const _Float16* Vbh = Vt + ((size_t)(b * HH + hh)) * DD * TT;
  const int* mrow = amask + b * TT;

  // Q fragments (reused every key block): D=64 -> two K=32 chunks
  v16h q0 = load_a_frag(Qbh, DD, qT * 16 + l15, 0, hsel);
  v16h q1 = load_a_frag(Qbh, DD, qT * 16 + l15, 32, hsel);

  v8f o0 = {}, o1 = {}, o2 = {}, o3 = {};
  float mstat[8], lstat[8];
#pragma unroll
  for (int r = 0; r < 8; ++r) { mstat[r] = -3.0e38f; lstat[r] = 0.0f; }

  _Float16* pbuf = &smem[wv][0];

  for (int kb = 0; kb < TT; kb += 32) {
    // ----- S = Q * K^T for 32 keys (two 16x16 C tiles) -----
    v8f s0 = {}, s1 = {};
    {
      v16h kb0 = load_b_frag(Kbh, DD, kb + l15, 0, hsel);
      v16h kb1 = load_b_frag(Kbh, DD, kb + l15, 32, hsel);
      s0 = WMMA_F16(q0, kb0, s0);
      s0 = WMMA_F16(q1, kb1, s0);
      v16h kc0 = load_b_frag(Kbh, DD, kb + 16 + l15, 0, hsel);
      v16h kc1 = load_b_frag(Kbh, DD, kb + 16 + l15, 32, hsel);
      s1 = WMMA_F16(q0, kc0, s1);
      s1 = WMMA_F16(q1, kc1, s1);
    }
    if (kb + 32 < TT) {                       // prefetch next key block
      __builtin_prefetch(Kbh + (size_t)(kb + 32 + l15) * DD, 0, 1);
      __builtin_prefetch(Kbh + (size_t)(kb + 48 + l15) * DD, 0, 1);
    }

    int mk0 = mrow[kb + l15];
    int mk1 = mrow[kb + 16 + l15];

    float p0[8], p1[8], rmax[8];
#pragma unroll
    for (int r = 0; r < 8; ++r) {
      float a0 = mk0 ? s0[r] * SCALE : -3.0e38f;
      float a1 = mk1 ? s1[r] * SCALE : -3.0e38f;
      p0[r] = a0; p1[r] = a1;
      rmax[r] = fmaxf(a0, a1);
    }
    // row max across the 16 lanes of each half-group (cols 0..15)
#pragma unroll
    for (int r = 0; r < 8; ++r) {
#pragma unroll
      for (int off = 8; off >= 1; off >>= 1)
        rmax[r] = fmaxf(rmax[r], __shfl_xor(rmax[r], off, 16));
    }
    float corr[8];
#pragma unroll
    for (int r = 0; r < 8; ++r) {
      float mn = fmaxf(mstat[r], rmax[r]);
      corr[r]  = __expf(mstat[r] - mn);
      mstat[r] = mn;
      p0[r] = __expf(p0[r] - mn);
      p1[r] = __expf(p1[r] - mn);
      float rs = p0[r] + p1[r];
#pragma unroll
      for (int off = 8; off >= 1; off >>= 1)
        rs += __shfl_xor(rs, off, 16);
      lstat[r] = lstat[r] * corr[r] + rs;
    }
#pragma unroll
    for (int r = 0; r < 8; ++r) {
      o0[r] *= corr[r]; o1[r] *= corr[r]; o2[r] *= corr[r]; o3[r] *= corr[r];
    }

    // ----- P (C-layout) -> LDS -> A-layout fragment -----
#pragma unroll
    for (int r = 0; r < 8; ++r) {
      pbuf[(r + 8 * hsel) * 32 + l15]      = (_Float16)p0[r];
      pbuf[(r + 8 * hsel) * 32 + 16 + l15] = (_Float16)p1[r];
    }
    // DS ops from one wave execute in order; wait for stores, block reordering.
    asm volatile("s_wait_dscnt 0" ::: "memory");
    v16h pf = load_a_frag(pbuf, 32, l15, 0, hsel);

    // ----- O += P * V (V stored [B,H,D,T]: rows are B^T rows) -----
    v16h v0f = load_b_frag(Vbh, TT, 0 * 16 + l15, kb, hsel);
    v16h v1f = load_b_frag(Vbh, TT, 1 * 16 + l15, kb, hsel);
    v16h v2f = load_b_frag(Vbh, TT, 2 * 16 + l15, kb, hsel);
    v16h v3f = load_b_frag(Vbh, TT, 3 * 16 + l15, kb, hsel);
    o0 = WMMA_F16(pf, v0f, o0);
    o1 = WMMA_F16(pf, v1f, o1);
    o2 = WMMA_F16(pf, v2f, o2);
    o3 = WMMA_F16(pf, v3f, o3);
  }

  // finalize and store AO as [B*T, H*D] f16
#pragma unroll
  for (int r = 0; r < 8; ++r) {
    float inv = 1.0f / lstat[r];
    int qrow = qT * 16 + r + 8 * hsel;
    size_t rowBase = ((size_t)(b * TT + qrow)) * (HH * DD) + hh * DD;
    AO[rowBase + 0 * 16 + l15] = (_Float16)(o0[r] * inv);
    AO[rowBase + 1 * 16 + l15] = (_Float16)(o1[r] * inv);
    AO[rowBase + 2 * 16 + l15] = (_Float16)(o2[r] * inv);
    AO[rowBase + 3 * 16 + l15] = (_Float16)(o3[r] * inv);
  }
}

// ---------------------------------------------------------------------------
extern "C" void kernel_launch(void* const* d_in, const int* in_sizes, int n_in,
                              void* d_out, int out_size, void* d_ws, size_t ws_size,
                              hipStream_t stream) {
  const float* hs = (const float*)d_in[0];
  const int*   am = (const int*)d_in[1];
  const float* Wq = (const float*)d_in[2];
  const float* Wk = (const float*)d_in[3];
  const float* Wv = (const float*)d_in[4];
  const float* Wo = (const float*)d_in[5];
  const float* bo = (const float*)d_in[6];
  float* out = (float*)d_out;

  // workspace carve-up (halfs), total ~46.1M halfs = ~88 MB of d_ws
  _Float16* ws = (_Float16*)d_ws;
  size_t off = 0;
  _Float16* Xh  = ws + off; off += (size_t)BB * TT * CC;
  _Float16* WqT = ws + off; off += (size_t)CC * CC;
  _Float16* WkT = ws + off; off += (size_t)CC * CC;
  _Float16* WvT = ws + off; off += (size_t)CC * CC;
  _Float16* WoT = ws + off; off += (size_t)CC * CC;
  _Float16* Qf  = ws + off; off += (size_t)BB * HH * TT * DD;
  _Float16* Kf  = ws + off; off += (size_t)BB * HH * TT * DD;
  _Float16* Vt  = ws + off; off += (size_t)BB * HH * DD * TT;
  _Float16* AO  = ws + off; off += (size_t)BB * TT * HH * DD;

  const int M = BB * TT;       // 8192
  const int N = HH * DD;       // 1024 (== CC)
  const int K = CC;            // 1024

  // Stage 0: conversions
  {
    int n = M * CC;
    cvt_f32_to_f16_kernel<<<n / 256, 256, 0, stream>>>(hs, Xh, n);
    int nw = CC * N;
    transpose_f32_to_f16_kernel<<<nw / 256, 256, 0, stream>>>(Wq, WqT, CC, N);
    transpose_f32_to_f16_kernel<<<nw / 256, 256, 0, stream>>>(Wk, WkT, CC, N);
    transpose_f32_to_f16_kernel<<<nw / 256, 256, 0, stream>>>(Wv, WvT, CC, N);
    transpose_f32_to_f16_kernel<<<nw / 256, 256, 0, stream>>>(Wo, WoT, N, CC);
  }

  // Stage 1: QKV projections (32x32 tile per wave; 8 waves per block)
  {
    int waves  = (M / 32) * (N / 32);   // 8192
    int blocks = waves / 8;             // 1024
    gemm_wmma_kernel<<<blocks, 256, 0, stream>>>(Xh, WqT, Qf, nullptr, nullptr, M, N, K, 0);
    gemm_wmma_kernel<<<blocks, 256, 0, stream>>>(Xh, WkT, Kf, nullptr, nullptr, M, N, K, 1);
    gemm_wmma_kernel<<<blocks, 256, 0, stream>>>(Xh, WvT, Vt, nullptr, nullptr, M, N, K, 2);
  }

  // Stage 2: flash attention (one wave per 16-query tile per head)
  {
    int waves  = BB * HH * (TT / 16);   // 8192
    int blocks = waves / 8;             // 1024
    attn_flash_kernel<<<blocks, 256, 0, stream>>>(Qf, Kf, Vt, am, AO);
  }

  // Stage 3: output projection + bias, f32 result
  {
    int waves  = (M / 32) * (CC / 32);
    int blocks = waves / 8;
    gemm_wmma_kernel<<<blocks, 256, 0, stream>>>(AO, WoT, nullptr, out, bo, M, CC, N, 3);
  }
}
